// MultiHeadAttention_31817117729497
// MI455X (gfx1250) — matrix operand order repaired and checked by reading, not compiled, not run on hardware
//
#include <hip/hip_runtime.h>
#include <math.h>

#define DEV __device__ __forceinline__

typedef __attribute__((ext_vector_type(16))) __bf16 v16bf;
typedef __attribute__((ext_vector_type(8)))  float  v8f;

constexpr int B_  = 2;
constexpr int S_  = 2048;
constexpr int D_  = 1024;
constexpr int H_  = 16;
constexpr int DH_ = 64;
constexpr int FF_ = 4096;
constexpr int NROW_ = B_ * S_;          // 4096 token rows
constexpr float SM_SCALE = 0.125f;      // 1/sqrt(DH)

// ---------- bf16 helpers (bit-exact storage as ushort) ----------
DEV unsigned short f2bf(float f) {
  unsigned int u = __float_as_uint(f);
  u += 0x7FFFu + ((u >> 16) & 1u);      // round-to-nearest-even
  return (unsigned short)(u >> 16);
}
DEV float bf2f(unsigned short u) { return __uint_as_float((unsigned int)u << 16); }
DEV __bf16 bfc(unsigned short u) { return __builtin_bit_cast(__bf16, u); }

// ---------- WMMA fragment loaders (wave32; contiguous k order: lane half h holds k 16h..16h+15) ----------
// A: 16x32 bf16 tile, row-major LDS tile at (row0, k0), leading dim ld.
// Per lane: two contiguous 8-ushort runs -> 2x ds_load_b128.
DEV v16bf ld_frag_a(const unsigned short* t, int ld) {
  const int lane = threadIdx.x & 31;
  const int m  = lane & 15;
  const int kh = (lane & 16) ? 16 : 0;  // lanes 16-31 hold K16-31
  v16bf a;
#pragma unroll
  for (int e = 0; e < 16; ++e) {
    const int k = kh + e;                     // e0-15 -> K kh+0..kh+15
    a[e] = bfc(t[m * ld + k]);
  }
  return a;
}
// B: 32x16 (K x N) fragment read from a TRANSPOSED LDS tile stored [N][K].
// Per lane: 16 contiguous ushorts at a 16B-aligned address -> 2x ds_load_b128.
DEV v16bf ld_frag_bt(const unsigned short* t, int ld) {
  const int lane = threadIdx.x & 31;
  const int n  = lane & 15;
  const int kb = (lane & 16) ? 16 : 0;  // lanes 16-31 hold K16-31
  v16bf b;
#pragma unroll
  for (int e = 0; e < 16; ++e) b[e] = bfc(t[n * ld + kb + e]);
  return b;
}

#define WMMA_BF16(A, Bf, C) \
  __builtin_amdgcn_wmma_f32_16x16x32_bf16(false, (A), false, (Bf), (short)0, (C), false, false)

#define V8F_ZERO {0.f,0.f,0.f,0.f,0.f,0.f,0.f,0.f}

DEV float gelu_f(float v) { return 0.5f * v * (1.0f + erff(v * 0.70710678118654752f)); }

// ---------- 64x64 block-tile GEMM mainloop ----------
// B is given TRANSPOSED in global memory: Bt_g[n][k] with row stride = K.
DEV void gemm_mainloop64(const unsigned short* __restrict__ A, size_t lda,   // A at (row0, 0)
                         const unsigned short* __restrict__ Btg,             // Bt at (col0, 0), ld = K
                         int K, int tid, int m0a, int m0b, int n0,
                         v8f& acc0, v8f& acc1,
                         unsigned short As[64][40], unsigned short Bt[64][40])
{
  for (int k0 = 0; k0 < K; k0 += 32) {
    { // stage A: 64x32, 8 contiguous ushort per thread
      const int r = tid >> 2, c = (tid & 3) * 8;
      const unsigned short* src = A + (size_t)r * lda + k0 + c;
#pragma unroll
      for (int i = 0; i < 8; ++i) As[r][c + i] = src[i];
    }
    { // stage Bt: 64x32, contiguous (already transposed in global)
      const int n = tid >> 2, c = (tid & 3) * 8;
      const unsigned short* src = Btg + (size_t)n * K + k0 + c;
#pragma unroll
      for (int i = 0; i < 8; ++i) Bt[n][c + i] = src[i];
    }
    __syncthreads();
    const v16bf bfr = ld_frag_bt(&Bt[n0][0], 40);
    const v16bf a0  = ld_frag_a(&As[m0a][0], 40);
    const v16bf a1  = ld_frag_a(&As[m0b][0], 40);
    acc0 = WMMA_BF16(a0, bfr, acc0);
    acc1 = WMMA_BF16(a1, bfr, acc1);
    __syncthreads();
  }
}

// ---------- 128x128 block-tile GEMM mainloop (8 waves, 2x4 subtiles/wave) ----------
DEV void gemm_mainloop128(const unsigned short* __restrict__ A, size_t lda,   // A at (row0, 0)
                          const unsigned short* __restrict__ Btg,             // Bt at (col0, 0), ld = K
                          int K, int tid, int wr, int wc,
                          v8f acc[2][4],
                          unsigned short As[128][40], unsigned short Bt[128][40])
{
  for (int k0 = 0; k0 < K; k0 += 32) {
    { // stage A: 128x32, 16 contiguous ushort per thread
      const int r = tid >> 1, c = (tid & 1) * 16;
      const unsigned short* src = A + (size_t)r * lda + k0 + c;
#pragma unroll
      for (int i = 0; i < 16; ++i) As[r][c + i] = src[i];
    }
    { // stage Bt: 128x32, contiguous
      const int n = tid >> 1, c = (tid & 1) * 16;
      const unsigned short* src = Btg + (size_t)n * K + k0 + c;
#pragma unroll
      for (int i = 0; i < 16; ++i) Bt[n][c + i] = src[i];
    }
    __syncthreads();
    v16bf bfr[4], af[2];
#pragma unroll
    for (int j = 0; j < 4; ++j) bfr[j] = ld_frag_bt(&Bt[(wc * 4 + j) * 16][0], 40);
#pragma unroll
    for (int i = 0; i < 2; ++i) af[i] = ld_frag_a(&As[(wr * 2 + i) * 16][0], 40);
#pragma unroll
    for (int i = 0; i < 2; ++i)
#pragma unroll
      for (int j = 0; j < 4; ++j)
        acc[i][j] = WMMA_BF16(af[i], bfr[j], acc[i][j]);
    __syncthreads();
  }
}

// ---------- batched f32 [R,C] -> bf16 transposed [C,R] (32x32 LDS tiles) ----------
__global__ __launch_bounds__(256)
void cvt_t_kernel(const float* __restrict__ in, unsigned short* __restrict__ out,
                  int R, int C) {
  __shared__ float t[32][33];
  const int batch = blockIdx.z;
  const float* src = in + (size_t)batch * R * C;
  unsigned short* dst = out + (size_t)batch * R * C;
  const int c0 = blockIdx.x * 32, r0 = blockIdx.y * 32;
  const int tx = threadIdx.x & 31, ty = threadIdx.x >> 5;   // 32 x 8
#pragma unroll
  for (int k = 0; k < 4; ++k) {
    const int r = ty + k * 8;
    t[r][tx] = src[(size_t)(r0 + r) * C + c0 + tx];
  }
  __syncthreads();
#pragma unroll
  for (int k = 0; k < 4; ++k) {
    const int c = ty + k * 8;
    dst[(size_t)(c0 + c) * R + r0 + tx] = f2bf(t[tx][c]);
  }
}

// ---------- LayerNorm (one block per token row), bf16 output ----------
__global__ __launch_bounds__(256)
void ln_kernel(const float* __restrict__ x, const float* __restrict__ g,
               const float* __restrict__ beta, unsigned short* __restrict__ out)
{
  __shared__ float red[256];
  const int row = blockIdx.x;
  const float* xr = x + (size_t)row * D_;
  float s = 0.f;
  for (int c = threadIdx.x; c < D_; c += 256) s += xr[c];
  red[threadIdx.x] = s; __syncthreads();
  for (int st = 128; st > 0; st >>= 1) {
    if (threadIdx.x < st) red[threadIdx.x] += red[threadIdx.x + st];
    __syncthreads();
  }
  const float mu = red[0] * (1.0f / D_);
  __syncthreads();
  float v = 0.f;
  for (int c = threadIdx.x; c < D_; c += 256) { const float d = xr[c] - mu; v += d * d; }
  red[threadIdx.x] = v; __syncthreads();
  for (int st = 128; st > 0; st >>= 1) {
    if (threadIdx.x < st) red[threadIdx.x] += red[threadIdx.x + st];
    __syncthreads();
  }
  const float rstd = rsqrtf(red[0] * (1.0f / D_) + 1e-5f);
  for (int c = threadIdx.x; c < D_; c += 256)
    out[(size_t)row * D_ + c] = f2bf((xr[c] - mu) * rstd * g[c] + beta[c]);
}

// ---------- per-head QKV projection: [S,D] x [DH,D]^T + bias ----------
// TROUT=false: write [B,H,S,DH] (Q and K).  TROUT=true: write [B,H,DH,S] (V, pre-transposed for flash).
template <bool TROUT>
__global__ __launch_bounds__(256)
void qkv_gemm_kernel(const unsigned short* __restrict__ Abf,   // [B*S, D] (ln1 out)
                     const unsigned short* __restrict__ Wtbf,  // [H, DH, D] transposed weights
                     const float* __restrict__ bias,           // [H, DH]
                     unsigned short* __restrict__ Obf)
{
  __shared__ unsigned short As[64][40];
  __shared__ unsigned short Bt[64][40];
  const int bh = blockIdx.y, b = bh / H_, h = bh % H_;
  const int s0 = blockIdx.x * 64;
  const int tid = threadIdx.x, wave = tid >> 5;
  const int m0a = (wave >> 2) * 16, m0b = ((wave >> 2) + 2) * 16, n0 = (wave & 3) * 16;
  v8f acc0 = V8F_ZERO, acc1 = V8F_ZERO;
  gemm_mainloop64(Abf + (size_t)(b * S_ + s0) * D_, D_,
                  Wtbf + (size_t)h * DH_ * D_,
                  D_, tid, m0a, m0b, n0, acc0, acc1, As, Bt);
  const int lane = tid & 31, n = lane & 15, mb = (lane & 16) ? 8 : 0;
  const float* bh_bias = bias + h * DH_;
#pragma unroll
  for (int r = 0; r < 8; ++r) {
    const int nn = n0 + n;
    const int mA = m0a + mb + r, mB = m0b + mb + r;
    const unsigned short vA = f2bf(acc0[r] + bh_bias[nn]);
    const unsigned short vB = f2bf(acc1[r] + bh_bias[nn]);
    if (TROUT) {  // [bh][dh][s]
      unsigned short* Ot = Obf + ((size_t)bh * DH_ + nn) * S_ + s0;
      Ot[mA] = vA;
      Ot[mB] = vB;
    } else {      // [bh][s][dh]
      unsigned short* Orow = Obf + ((size_t)bh * S_ + s0) * DH_;
      Orow[(size_t)mA * DH_ + nn] = vA;
      Orow[(size_t)mB * DH_ + nn] = vB;
    }
  }
}

// ---------- causal flash attention, one block per (q-tile of 64, b*H+h) ----------
// Q,K in [B,H,S,DH]; V pre-transposed in [B,H,DH,S].
// Epilogue writes d_out[b,s,h*DH+dh] = x + attn_out (head slices disjoint).
__global__ __launch_bounds__(256)
void flash_attn_kernel(const unsigned short* __restrict__ Q,
                       const unsigned short* __restrict__ K,
                       const unsigned short* __restrict__ Vt_g,
                       const float* __restrict__ x,
                       float* __restrict__ out)
{
  __shared__ unsigned short Qs[64][72];   // [q][dh]    (A for QK^T; pre-scaled by 1/sqrt(DH))
  __shared__ unsigned short Ks[64][72];   // [key][dh]  (transposed-B for QK^T: natural layout)
  __shared__ unsigned short Vt[64][72];   // [dh][key]  (transposed-B for P*V: pre-transposed global)
  __shared__ float          Ss[64][65];   // f32 scores
  __shared__ unsigned short Ps[64][72];   // bf16 probs (A for P*V)
  __shared__ float mrow[64], lrow[64], arow[64];

  const int bh = blockIdx.y, b = bh / H_, h = bh % H_;
  const int qt = blockIdx.x, s0 = qt * 64;
  const int tid = threadIdx.x, wave = tid >> 5, lane = tid & 31;
  const int m0a = (wave >> 2) * 16, m0b = ((wave >> 2) + 2) * 16, n0 = (wave & 3) * 16;
  const int cn = lane & 15, cmb = (lane & 16) ? 8 : 0;

  const unsigned short* Qb = Q    + (size_t)bh * S_ * DH_;
  const unsigned short* Kb = K    + (size_t)bh * S_ * DH_;
  const unsigned short* Vb = Vt_g + (size_t)bh * DH_ * S_;   // [dh][s]

  { // stage Q tile (64x64), folding in the softmax scale
    const int r = tid >> 2, c = (tid & 3) * 16;
    const unsigned short* src = Qb + (size_t)(s0 + r) * DH_ + c;
#pragma unroll
    for (int i = 0; i < 16; ++i) Qs[r][c + i] = f2bf(bf2f(src[i]) * SM_SCALE);
  }
  if (tid < 64) { mrow[tid] = -1e30f; lrow[tid] = 0.f; }
  v8f o0 = V8F_ZERO, o1 = V8F_ZERO;
  __syncthreads();

  for (int kt = 0; kt <= qt; ++kt) {
    const int k0g = kt * 64;
    { // stage K [key][dh] and V [dh][key] — both contiguous copies
      const int r = tid >> 2, c = (tid & 3) * 16;
      const unsigned short* ks = Kb + (size_t)(k0g + r) * DH_ + c;
      const unsigned short* vs = Vb + (size_t)r * S_ + k0g + c;
#pragma unroll
      for (int i = 0; i < 16; ++i) { Ks[r][c + i] = ks[i]; Vt[r][c + i] = vs[i]; }
    }
    __syncthreads();

    // ---- scores = (Q/sqrt(DH)) * K^T ----
    v8f sa = V8F_ZERO, sb = V8F_ZERO;
#pragma unroll
    for (int ks = 0; ks < DH_; ks += 32) {
      const v16bf bfr = ld_frag_bt(&Ks[n0][ks], 72);
      const v16bf a0  = ld_frag_a(&Qs[m0a][ks], 72);
      const v16bf a1  = ld_frag_a(&Qs[m0b][ks], 72);
      sa = WMMA_BF16(a0, bfr, sa);
      sb = WMMA_BF16(a1, bfr, sb);
    }
    // write masked scores to LDS
#pragma unroll
    for (int r = 0; r < 8; ++r) {
      const int mA = m0a + cmb + r, mB = m0b + cmb + r, nn = n0 + cn;
      float v0 = sa[r], v1 = sb[r];
      if (k0g + nn > s0 + mA) v0 = -1e30f;   // causal mask
      if (k0g + nn > s0 + mB) v1 = -1e30f;
      Ss[mA][nn] = v0;
      Ss[mB][nn] = v1;
    }
    __syncthreads();

    // ---- online softmax row update (one thread per row) ----
    if (tid < 64) {
      const float mold = mrow[tid];
      float mx = mold;
      for (int c = 0; c < 64; ++c) mx = fmaxf(mx, Ss[tid][c]);
      const float al = __expf(mold - mx);
      float sum = 0.f;
      for (int c = 0; c < 64; ++c) {
        const float p = __expf(Ss[tid][c] - mx);
        sum += p;
        Ps[tid][c] = f2bf(p);
      }
      mrow[tid] = mx;
      lrow[tid] = lrow[tid] * al + sum;
      arow[tid] = al;
    }
    __syncthreads();

    // ---- rescale O, then O += P * V ----
#pragma unroll
    for (int r = 0; r < 8; ++r) {
      o0[r] *= arow[m0a + cmb + r];
      o1[r] *= arow[m0b + cmb + r];
    }
#pragma unroll
    for (int ks = 0; ks < 64; ks += 32) {
      const v16bf bfr = ld_frag_bt(&Vt[n0][ks], 72);
      const v16bf a0  = ld_frag_a(&Ps[m0a][ks], 72);
      const v16bf a1  = ld_frag_a(&Ps[m0b][ks], 72);
      o0 = WMMA_BF16(a0, bfr, o0);
      o1 = WMMA_BF16(a1, bfr, o1);
    }
    __syncthreads();
  }

  // epilogue: residual add, normalize by l
  const float* xb = x   + ((size_t)b * S_ + s0) * D_ + h * DH_;
  float*       ob = out + ((size_t)b * S_ + s0) * D_ + h * DH_;
#pragma unroll
  for (int r = 0; r < 8; ++r) {
    const int mA = m0a + cmb + r, mB = m0b + cmb + r, nn = n0 + cn;
    ob[(size_t)mA * D_ + nn] = xb[(size_t)mA * D_ + nn] + o0[r] / lrow[mA];
    ob[(size_t)mB * D_ + nn] = xb[(size_t)mB * D_ + nn] + o1[r] / lrow[mB];
  }
}

// ---------- FFN GEMM 1: [4096,D] x W1 + b1, GELU -> bf16 (128x128 tiles) ----------
__global__ __launch_bounds__(256)
void ffn1_kernel(const unsigned short* __restrict__ A,    // [NROW, D] (ln2 out)
                 const unsigned short* __restrict__ W1t,  // [FF, D] transposed
                 const float* __restrict__ bias,          // [FF]
                 unsigned short* __restrict__ Out)        // [NROW, FF] bf16
{
  __shared__ unsigned short As[128][40];
  __shared__ unsigned short Bt[128][40];
  const int tid = threadIdx.x, wave = tid >> 5;
  const int wr = wave >> 1, wc = wave & 1;
  const int row0 = blockIdx.x * 128, col0 = blockIdx.y * 128;
  v8f acc[2][4] = {{V8F_ZERO, V8F_ZERO, V8F_ZERO, V8F_ZERO},
                   {V8F_ZERO, V8F_ZERO, V8F_ZERO, V8F_ZERO}};
  gemm_mainloop128(A + (size_t)row0 * D_, D_, W1t + (size_t)col0 * D_,
                   D_, tid, wr, wc, acc, As, Bt);
  const int lane = tid & 31, n = lane & 15, mb = (lane & 16) ? 8 : 0;
#pragma unroll
  for (int i = 0; i < 2; ++i)
#pragma unroll
    for (int j = 0; j < 4; ++j)
#pragma unroll
      for (int r = 0; r < 8; ++r) {
        const int row = row0 + (wr * 2 + i) * 16 + mb + r;
        const int nn  = col0 + (wc * 4 + j) * 16 + n;
        Out[(size_t)row * FF_ + nn] = f2bf(gelu_f(acc[i][j][r] + bias[nn]));
      }
}

// ---------- FFN GEMM 2: [4096,FF] x W2 + b2 + residual -> out f32 (128x128 tiles) ----------
__global__ __launch_bounds__(256)
void ffn2_kernel(const unsigned short* __restrict__ A,    // [NROW, FF] bf16
                 const unsigned short* __restrict__ W2t,  // [D, FF] transposed
                 const float* __restrict__ bias,          // [D]
                 float* __restrict__ out)                 // [NROW, D] (holds x+attn, updated in place)
{
  __shared__ unsigned short As[128][40];
  __shared__ unsigned short Bt[128][40];
  const int tid = threadIdx.x, wave = tid >> 5;
  const int wr = wave >> 1, wc = wave & 1;
  const int row0 = blockIdx.x * 128, col0 = blockIdx.y * 128;
  v8f acc[2][4] = {{V8F_ZERO, V8F_ZERO, V8F_ZERO, V8F_ZERO},
                   {V8F_ZERO, V8F_ZERO, V8F_ZERO, V8F_ZERO}};
  gemm_mainloop128(A + (size_t)row0 * FF_, FF_, W2t + (size_t)col0 * FF_,
                   FF_, tid, wr, wc, acc, As, Bt);
  const int lane = tid & 31, n = lane & 15, mb = (lane & 16) ? 8 : 0;
#pragma unroll
  for (int i = 0; i < 2; ++i)
#pragma unroll
    for (int j = 0; j < 4; ++j)
#pragma unroll
      for (int r = 0; r < 8; ++r) {
        const int row = row0 + (wr * 2 + i) * 16 + mb + r;
        const int nn  = col0 + (wc * 4 + j) * 16 + n;
        const size_t idx = (size_t)row * D_ + nn;
        out[idx] = out[idx] + acc[i][j][r] + bias[nn];
      }
}

// ======================= launcher =======================
extern "C" void kernel_launch(void* const* d_in, const int* in_sizes, int n_in,
                              void* d_out, int out_size, void* d_ws, size_t ws_size,
                              hipStream_t stream)
{
  const float* x   = (const float*)d_in[0];
  const float* Wq  = (const float*)d_in[1];
  const float* bq  = (const float*)d_in[2];
  const float* Wk  = (const float*)d_in[3];
  const float* bk  = (const float*)d_in[4];
  const float* Wv  = (const float*)d_in[5];
  const float* bv  = (const float*)d_in[6];
  const float* g1  = (const float*)d_in[7];
  const float* be1 = (const float*)d_in[8];
  const float* g2  = (const float*)d_in[9];
  const float* be2 = (const float*)d_in[10];
  const float* W1  = (const float*)d_in[11];
  const float* b1  = (const float*)d_in[12];
  const float* W2  = (const float*)d_in[13];
  const float* b2  = (const float*)d_in[14];
  float* out = (float*)d_out;

  // workspace carve-up (bf16 stored as ushort)
  char* ws = (char*)d_ws;
  size_t off = 0;
  auto alloc = [&](size_t bytes) -> void* {
    void* p = ws + off;
    off = (off + bytes + 255) & ~(size_t)255;
    return p;
  };
  unsigned short* hbf  = (unsigned short*)alloc((size_t)NROW_ * D_ * 2);     // ln1 out
  unsigned short* h2bf = (unsigned short*)alloc((size_t)NROW_ * D_ * 2);     // ln2 out
  unsigned short* wqt  = (unsigned short*)alloc((size_t)H_ * D_ * DH_ * 2);  // [H,DH,D]
  unsigned short* wkt  = (unsigned short*)alloc((size_t)H_ * D_ * DH_ * 2);
  unsigned short* wvt  = (unsigned short*)alloc((size_t)H_ * D_ * DH_ * 2);
  unsigned short* qbf  = (unsigned short*)alloc((size_t)B_ * H_ * S_ * DH_ * 2);  // [B,H,S,DH]
  unsigned short* kbf  = (unsigned short*)alloc((size_t)B_ * H_ * S_ * DH_ * 2);  // [B,H,S,DH]
  unsigned short* vtbf = (unsigned short*)alloc((size_t)B_ * H_ * DH_ * S_ * 2);  // [B,H,DH,S]
  unsigned short* w1t  = (unsigned short*)alloc((size_t)D_ * FF_ * 2);            // [FF,D]
  unsigned short* w2t  = (unsigned short*)alloc((size_t)FF_ * D_ * 2);            // [D,FF]
  unsigned short* fbf  = (unsigned short*)alloc((size_t)NROW_ * FF_ * 2);         // [NROW,FF]

  // 1) weights -> bf16, transposed to [N][K]
  cvt_t_kernel<<<dim3(DH_ / 32, D_ / 32, H_), 256, 0, stream>>>(Wq, wqt, D_, DH_);
  cvt_t_kernel<<<dim3(DH_ / 32, D_ / 32, H_), 256, 0, stream>>>(Wk, wkt, D_, DH_);
  cvt_t_kernel<<<dim3(DH_ / 32, D_ / 32, H_), 256, 0, stream>>>(Wv, wvt, D_, DH_);
  cvt_t_kernel<<<dim3(FF_ / 32, D_ / 32, 1), 256, 0, stream>>>(W1, w1t, D_, FF_);
  cvt_t_kernel<<<dim3(D_ / 32, FF_ / 32, 1), 256, 0, stream>>>(W2, w2t, FF_, D_);

  // 2) LN1
  ln_kernel<<<NROW_, 256, 0, stream>>>(x, g1, be1, hbf);

  // 3) QKV projections (V written pre-transposed)
  dim3 qkv_grid(S_ / 64, B_ * H_);
  qkv_gemm_kernel<false><<<qkv_grid, 256, 0, stream>>>(hbf, wqt, bq, qbf);
  qkv_gemm_kernel<false><<<qkv_grid, 256, 0, stream>>>(hbf, wkt, bk, kbf);
  qkv_gemm_kernel<true ><<<qkv_grid, 256, 0, stream>>>(hbf, wvt, bv, vtbf);

  // 4) causal flash attention, writes out = x + attn
  flash_attn_kernel<<<dim3(S_ / 64, B_ * H_), 256, 0, stream>>>(qbf, kbf, vtbf, x, out);

  // 5) LN2
  ln_kernel<<<NROW_, 256, 0, stream>>>(out, g2, be2, h2bf);

  // 6) FFN (128x128 tiles)
  ffn1_kernel<<<dim3(NROW_ / 128, FF_ / 128), 256, 0, stream>>>(h2bf, w1t, b1, fbf);
  ffn2_kernel<<<dim3(NROW_ / 128, D_ / 128), 256, 0, stream>>>(fbf, w2t, b2, out);
}
